// CondConv2d_35699768164364
// MI455X (gfx1250) — matrix operand order repaired and checked by reading, not compiled, not run on hardware
//
#include <hip/hip_runtime.h>

typedef __attribute__((ext_vector_type(16))) __bf16 v16bf;
typedef __attribute__((ext_vector_type(8)))  __bf16 v8bf;
typedef __attribute__((ext_vector_type(8)))  float  v8f;

#define NB 8
#define CIN 64
#define COUT 64
#define HH 128
#define WW 128
#define KEXP 4
#define HWB (HH*WW)                 // 16384
#define IMG (CIN*HWB)               // 1048576 elements per sample
#define TOTROWS (NB*CIN*HH)         // 65536 rows of length 128
#define WEFF_ELEMS (NB*COUT*9*CIN)  // 294912 per hi/lo array

// ---- bf16 helpers (bit-exact, no reliance on native __bf16 conversions) ----
__device__ inline __bf16 f2bf(float f) {
  unsigned u = __builtin_bit_cast(unsigned, f);
  u = (u + 0x7FFFu + ((u >> 16) & 1u)) >> 16;   // RTNE
  return __builtin_bit_cast(__bf16, (unsigned short)u);
}
__device__ inline float bf2f(__bf16 b) {
  unsigned u = ((unsigned)__builtin_bit_cast(unsigned short, b)) << 16;
  return __builtin_bit_cast(float, u);
}

// =====================================================================
// A1: per-row (n,d,h) stats: RS = sum over w, F = x[...,0], L = x[...,127]
// one wave per 32 rows (one row at a time), float4 coalesced loads, shfl butterfly reduce
// =====================================================================
__global__ __launch_bounds__(256) void rowstats_kernel(
    const float* __restrict__ x, float* __restrict__ RS,
    float* __restrict__ F, float* __restrict__ L) {
  const int gw   = blockIdx.x * 8 + (threadIdx.x >> 5);
  const int lane = threadIdx.x & 31;
  const int r0 = gw * 32;
  for (int i = 0; i < 32; ++i) {
    const int r = r0 + i;
    const float4 v = ((const float4*)(x + (size_t)r * WW))[lane];
    float s = v.x + v.y + v.z + v.w;
    #pragma unroll
    for (int off = 16; off > 0; off >>= 1) s += __shfl_xor(s, off);
    if (lane == 0)  { RS[r] = s; F[r] = v.x; }
    if (lane == 31) { L[r] = v.w; }
  }
}

// =====================================================================
// A2: per (n,d): reduce over h into 3 H-classes x {F, RS-F-L, L} -> T[n][d][9]
// one wave per (n,d)
// =====================================================================
__global__ __launch_bounds__(256) void hreduce_kernel(
    const float* __restrict__ RS, const float* __restrict__ F,
    const float* __restrict__ L, float* __restrict__ T) {
  const int gw   = blockIdx.x * 8 + (threadIdx.x >> 5); // 0..511 = (n,d)
  const int lane = threadIdx.x & 31;
  const int n = gw >> 6, d = gw & 63;
  const int base = (n * 64 + d) * 128;
  float a[9];
  #pragma unroll
  for (int i = 0; i < 9; ++i) a[i] = 0.f;
  #pragma unroll
  for (int j = 0; j < 4; ++j) {
    const int h = lane + 32 * j;
    const float f = F[base + h];
    const float l2 = L[base + h];
    const float m = RS[base + h] - f - l2;
    if (h == 0)        { a[0] += f; a[1] += m; a[2] += l2; }
    else if (h == 127) { a[6] += f; a[7] += m; a[8] += l2; }
    else               { a[3] += f; a[4] += m; a[5] += l2; }
  }
  #pragma unroll
  for (int i = 0; i < 9; ++i)
    for (int off = 16; off > 0; off >>= 1) a[i] += __shfl_xor(a[i], off);
  if (lane == 0) {
    #pragma unroll
    for (int i = 0; i < 9; ++i) T[(n * 64 + d) * 9 + i] = a[i];
  }
}

// =====================================================================
// A3: per sample: 45 region bins -> logits (analytic tap coverage) ->
//     softmax -> fused W_eff = conv_w + sum_k att_k*weight_k, split to bf16
//     hi/lo in layout [n][co][tap][ci]
// =====================================================================
__global__ __launch_bounds__(256) void logits_weff_kernel(
    const float* __restrict__ T,
    const float* __restrict__ n0w, const float* __restrict__ n0b,
    const float* __restrict__ n1w, const float* __restrict__ n1b,
    const float* __restrict__ n2w, const float* __restrict__ n2b,
    const float* __restrict__ conv_w, const float* __restrict__ weight,
    __bf16* __restrict__ WHi, __bf16* __restrict__ WLo) {
  __shared__ float bins[45];
  __shared__ float logacc[KEXP];
  __shared__ float att[KEXP];
  const int n = blockIdx.x;
  const int tid = threadIdx.x;

  if (tid < 45) {
    const int cd = tid / 9, rem = tid % 9;
    int dlo, dhi;
    if (cd == 0)      { dlo = 0;  dhi = 0;  }
    else if (cd == 1) { dlo = 1;  dhi = 1;  }
    else if (cd == 2) { dlo = 2;  dhi = 61; }
    else if (cd == 3) { dlo = 62; dhi = 62; }
    else              { dlo = 63; dhi = 63; }
    float s = 0.f;
    for (int d = dlo; d <= dhi; ++d) s += T[(n * 64 + d) * 9 + rem];
    bins[tid] = s;
  }
  if (tid < KEXP) logacc[tid] = 0.f;
  __syncthreads();

  if (tid < KEXP * 45) {
    const int k = tid / 45, j = tid % 45;
    const int cd = j / 9, rem = j % 9, ch = rem / 3, cw = rem % 3;
    float c = 0.f;
    // net1: 3x3x3, pad (1,1,1). Tap (dd,dh,dw) covers this class iff valid.
    for (int dd = 0; dd < 3; ++dd) {
      if ((cd == 0 && dd == 2) || (cd == 4 && dd == 0)) continue;
      for (int dh = 0; dh < 3; ++dh) {
        if ((ch == 0 && dh == 2) || (ch == 2 && dh == 0)) continue;
        for (int dw = 0; dw < 3; ++dw) {
          if ((cw == 0 && dw == 2) || (cw == 2 && dw == 0)) continue;
          c += n1w[k * 27 + (dd * 3 + dh) * 3 + dw];
        }
      }
    }
    // net2: 5x3x3, pad (2,1,1).
    for (int dd = 0; dd < 5; ++dd) {
      bool okd;
      if (cd == 0)      okd = (dd <= 2);
      else if (cd == 1) okd = (dd <= 3);
      else if (cd == 2) okd = true;
      else if (cd == 3) okd = (dd >= 1);
      else              okd = (dd >= 2);
      if (!okd) continue;
      for (int dh = 0; dh < 3; ++dh) {
        if ((ch == 0 && dh == 2) || (ch == 2 && dh == 0)) continue;
        for (int dw = 0; dw < 3; ++dw) {
          if ((cw == 0 && dw == 2) || (cw == 2 && dw == 0)) continue;
          c += n2w[k * 45 + (dd * 3 + dh) * 3 + dw];
        }
      }
    }
    atomicAdd(&logacc[k], c * bins[j]);
  }
  __syncthreads();

  if (tid == 0) {
    const float invM = 1.0f / (float)IMG;
    float tot = 0.f;
    for (int j = 0; j < 45; ++j) tot += bins[j];
    float lg[KEXP]; float mx = -1e30f;
    for (int k = 0; k < KEXP; ++k) {
      lg[k] = n0b[k] + n1b[k] + n2b[k] + n0w[k] * tot * invM + logacc[k] * invM;
      mx = fmaxf(mx, lg[k]);
    }
    float se = 0.f;
    for (int k = 0; k < KEXP; ++k) { lg[k] = __expf(lg[k] - mx); se += lg[k]; }
    for (int k = 0; k < KEXP; ++k) att[k] = lg[k] / se;
  }
  __syncthreads();

  // Fused effective weight, transposed to [co][tap][ci], bf16 hi/lo split
  for (int e = tid; e < COUT * 9 * CIN; e += 256) {
    const int co = e / 576, rem = e % 576, tap = rem >> 6, ci = rem & 63;
    float wv = conv_w[(co * 64 + ci) * 9 + tap];
    #pragma unroll
    for (int k = 0; k < KEXP; ++k)
      wv += att[k] * weight[((k * 64 + co) * 64 + ci) * 9 + tap];
    const __bf16 hi = f2bf(wv);
    const __bf16 lo = f2bf(wv - bf2f(hi));
    const int di = ((n * 64 + co) * 9 + tap) * 64 + ci;
    WHi[di] = hi; WLo[di] = lo;
  }
}

// =====================================================================
// B: fused per-sample 3x3 conv via WMMA implicit GEMM, bf16x3 split.
// Block: 16x16 output tile, 256 threads = 8 waves
//   wave -> (co-tile m0 = 16*(w>>1), pixel-half nh = 128*(w&1))
// LDS: 18x18 halo x 32-ci chunk, bf16 hi/lo, [pixel][ci] so B-frags are
// contiguous 32B per lane.
// =====================================================================
__global__ __launch_bounds__(256) void condconv_kernel(
    const float* __restrict__ x, const __bf16* __restrict__ WHi,
    const __bf16* __restrict__ WLo, const float* __restrict__ bias,
    float* __restrict__ out) {
  __shared__ __align__(32) __bf16 shHi[324 * 32];
  __shared__ __align__(32) __bf16 shLo[324 * 32];
  const int n  = blockIdx.y;
  const int t  = blockIdx.x;
  const int h0 = (t >> 3) * 16, w0 = (t & 7) * 16;
  const int tid  = threadIdx.x;
  const int lane = tid & 31;
  const int wv   = tid >> 5;
  const int m0 = (wv >> 1) * 16;     // co tile base
  const int nh = (wv & 1) * 128;     // pixel-half base (0 or 128)
  const bool hiHalf = lane >= 16;
  const int lc = lane & 15;

  v8f acc[8];
  const v8f vzero = {0.f, 0.f, 0.f, 0.f, 0.f, 0.f, 0.f, 0.f};
  #pragma unroll
  for (int i = 0; i < 8; ++i) acc[i] = vzero;

  const float* xn = x + (size_t)n * IMG;

  for (int c0 = 0; c0 < CIN; c0 += 32) {
    __syncthreads();
    // stage 18x18 halo x 32 channels, split to bf16 hi/lo, [pixel][ci]
    for (int e = tid; e < 324 * 32; e += 256) {
      const int c = e / 324;
      const int p = e - c * 324;
      const int py = p / 18, px = p - py * 18;
      const int h = h0 - 1 + py, w = w0 - 1 + px;
      float v = 0.f;
      if ((unsigned)h < 128u && (unsigned)w < 128u)
        v = xn[((c0 + c) << 14) + (h << 7) + w];
      const __bf16 hb = f2bf(v);
      shHi[p * 32 + c] = hb;
      shLo[p * 32 + c] = f2bf(v - bf2f(hb));
    }
    __syncthreads();

    for (int tap = 0; tap < 9; ++tap) {
      const int kh = tap / 3, kw = tap - kh * 3;
      // ---- A fragments (weights), contiguous k order, same as B:
      // lanes 0-15 hold K{0..15} for row M=lc; lanes 16-31 K{16..31}
      const int coA = m0 + lc;
      const size_t wbase =
          ((size_t)(n * 64 + coA) * 9 + tap) * 64 + c0 + (hiHalf ? 16 : 0);
      union { v16bf v; v8bf h[2]; } ahi, alo;
      ahi.h[0] = *(const v8bf*)(WHi + wbase);
      ahi.h[1] = *(const v8bf*)(WHi + wbase + 8);
      alo.h[0] = *(const v8bf*)(WLo + wbase);
      alo.h[1] = *(const v8bf*)(WLo + wbase + 8);

      #pragma unroll
      for (int nt = 0; nt < 8; ++nt) {
        // ---- B fragment (activations), 32x16 layout: lane half selects K 0-15 / 16-31
        const int p  = nh + nt * 16 + lc;       // output pixel (col N = lc)
        const int ph = p >> 4, pw = p & 15;
        const int lp = (ph + kh) * 18 + (pw + kw);
        const int soff = lp * 32 + (hiHalf ? 16 : 0);
        const v16bf bhi = *(const v16bf*)(shHi + soff);
        const v16bf blo = *(const v16bf*)(shLo + soff);
        acc[nt] = __builtin_amdgcn_wmma_f32_16x16x32_bf16(
            false, ahi.v, false, bhi, (short)0, acc[nt], false, false);
        acc[nt] = __builtin_amdgcn_wmma_f32_16x16x32_bf16(
            false, ahi.v, false, blo, (short)0, acc[nt], false, false);
        acc[nt] = __builtin_amdgcn_wmma_f32_16x16x32_bf16(
            false, alo.v, false, bhi, (short)0, acc[nt], false, false);
      }
    }
  }

  // Epilogue: C/D layout — lanes 0-15: M=r, N=lane; lanes 16-31: M=8+r, N=lane-16
  float* on = out + (size_t)n * IMG;
  const int coB = m0 + (hiHalf ? 8 : 0);
  #pragma unroll
  for (int nt = 0; nt < 8; ++nt) {
    const int p  = nh + nt * 16 + lc;
    const int ph = p >> 4, pw = p & 15;
    #pragma unroll
    for (int r = 0; r < 8; ++r) {
      const int co = coB + r;
      on[(co << 14) + ((h0 + ph) << 7) + (w0 + pw)] = acc[nt][r] + bias[co];
    }
  }
}

// =====================================================================
extern "C" void kernel_launch(void* const* d_in, const int* in_sizes, int n_in,
                              void* d_out, int out_size, void* d_ws, size_t ws_size,
                              hipStream_t stream) {
  const float* x      = (const float*)d_in[0];
  const float* weight = (const float*)d_in[1];
  const float* conv_w = (const float*)d_in[2];
  const float* conv_b = (const float*)d_in[3];
  const float* n0w    = (const float*)d_in[4];
  const float* n0b    = (const float*)d_in[5];
  const float* n1w    = (const float*)d_in[6];
  const float* n1b    = (const float*)d_in[7];
  const float* n2w    = (const float*)d_in[8];
  const float* n2b    = (const float*)d_in[9];
  float* out = (float*)d_out;

  char* ws = (char*)d_ws;                 // ~2 MB used
  float* RS = (float*)ws;                 // 65536 floats
  float* F  = RS + TOTROWS;               // 65536
  float* L  = F + TOTROWS;                // 65536
  float* T  = L + TOTROWS;                // 8*64*9 = 4608
  __bf16* WHi = (__bf16*)(ws + (size_t)(3 * TOTROWS + 4608) * sizeof(float));
  __bf16* WLo = WHi + WEFF_ELEMS;

  rowstats_kernel<<<256, 256, 0, stream>>>(x, RS, F, L);
  hreduce_kernel<<<64, 256, 0, stream>>>(RS, F, L, T);
  logits_weff_kernel<<<NB, 256, 0, stream>>>(T, n0w, n0b, n1w, n1b, n2w, n2b,
                                             conv_w, weight, WHi, WLo);
  condconv_kernel<<<dim3(64, NB), 256, 0, stream>>>(x, WHi, WLo, conv_b, out);
}
